// SparseDeformableAttentionBlock_11519102288658
// MI455X (gfx1250) — matrix operand order repaired and checked by reading, not compiled, not run on hardware
//
#include <hip/hip_runtime.h>
#include <hip/hip_bf16.h>

#define DD      256
#define HEADS   8
#define HDIM    32
#define LEVELS  4
#define POINTS  4
#define BB      2
#define NQ      8192
#define HMAX    128
#define WMAX    128
#define GK      256   // K dim for all GEMMs in this block
#define NB      128   // column block per workgroup (B panel = 256x128 bf16 = 64KB LDS)

typedef __attribute__((ext_vector_type(16))) __bf16 v16bf;
typedef __attribute__((ext_vector_type(8)))  float  v8f;

__device__ __forceinline__ __bf16 f32_to_bf16(float f) {
    unsigned u = __builtin_bit_cast(unsigned, f);
    unsigned r = u + 0x7FFFu + ((u >> 16) & 1u);   // round-to-nearest-even
    unsigned short hs = (unsigned short)(r >> 16);
    return __builtin_bit_cast(__bf16, hs);
}

// ---------------------------------------------------------------------------
// Kernel 1: q = LayerNorm(queries + pos) * gamma + beta.  One wave32 per row.
// ---------------------------------------------------------------------------
__global__ void ln_kernel(const float* __restrict__ qin,
                          const float* __restrict__ pos,
                          const float* __restrict__ gamma,
                          const float* __restrict__ beta,
                          float* __restrict__ qout, int nrows) {
    int wid  = (blockIdx.x * blockDim.x + threadIdx.x) >> 5;
    int lane = threadIdx.x & 31;
    if (wid >= nrows) return;
    const size_t base = (size_t)wid * DD;

    float x[DD / 32];
    float s = 0.f;
#pragma unroll
    for (int i = 0; i < DD / 32; ++i) {
        int e = lane + 32 * i;
        x[i] = qin[base + e] + pos[base + e];
        s += x[i];
    }
#pragma unroll
    for (int m = 16; m >= 1; m >>= 1) s += __shfl_xor(s, m, 32);
    float mean = s * (1.0f / DD);

    float v = 0.f;
#pragma unroll
    for (int i = 0; i < DD / 32; ++i) { float d = x[i] - mean; v += d * d; }
#pragma unroll
    for (int m = 16; m >= 1; m >>= 1) v += __shfl_xor(v, m, 32);
    float rstd = rsqrtf(v * (1.0f / DD) + 1e-5f);

#pragma unroll
    for (int i = 0; i < DD / 32; ++i) {
        int e = lane + 32 * i;
        qout[base + e] = (x[i] - mean) * rstd * gamma[e] + beta[e];
    }
}

// ---------------------------------------------------------------------------
// Kernel 2: row-major GEMM  C = A(MxK) * B(KxN) [+bias] [+residual], K = 256.
// Workgroup: 8 waves -> 128 rows x 128 cols of C.
// B panel (256 x NB) is converted to bf16 and staged TRANSPOSED in LDS with a
// 32B-chunk XOR swizzle (chunk' = chunk ^ (n&15)) so both staging stores and
// fragment loads spread across banks while every fragment read stays one
// contiguous 32B v16bf access.  Each A fragment (load + bf16 convert) is
// amortized over NB/16 = 8 v_wmma_f32_16x16x32_bf16 issues.
// ---------------------------------------------------------------------------
__global__ void __launch_bounds__(256)
wmma_gemm_bf16(const float* __restrict__ A,
               const float* __restrict__ Bm,
               const float* __restrict__ bias,
               const float* __restrict__ resid,
               float* __restrict__ C,
               int M, int N) {
    __shared__ __bf16 Blds[NB * GK];   // 64 KB, [n][k-swizzled]

    const int nblk  = N / NB;                       // 1 (N=128) or 2 (N=256)
    const int tileM = blockIdx.x / nblk;
    const int col0  = (blockIdx.x % nblk) * NB;
    const int wave  = threadIdx.x >> 5;
    const int lane  = threadIdx.x & 31;
    const int row0  = tileM * 128 + (wave << 4);

    // ---- stage B panel: 2048 chunks of 16 K-elements, bf16, swizzled ----
#pragma unroll
    for (int i = 0; i < 8; ++i) {
        int e  = i * 256 + threadIdx.x;   // chunk id: 128 n * 16 c
        int n  = e >> 4;
        int c  = e & 15;
        int ks = c << 4;
        v16bf t;
#pragma unroll
        for (int j = 0; j < 16; ++j)
            t[j] = f32_to_bf16(Bm[(size_t)(ks + j) * N + col0 + n]);
        *reinterpret_cast<v16bf*>(&Blds[n * GK + ((c ^ (n & 15)) << 4)]) = t;
    }
    __syncthreads();
    if (row0 >= M) return;

    // A fragment addressing, contiguous K order (B read the same way):
    //   lanes 0-15: row=lane,    K = {0..15} + kb
    //   lanes16-31: row=lane-16, K = {16..31} + kb
    const int ln16 = lane & 15;
    const int koff = (lane >> 4) << 4;   // A half-select: 0 or 16
    const int kofB = (lane >> 4) << 4;   // B K-range per lane: 0 or 16
    const float* arow = A + (size_t)(row0 + ln16) * GK;

    v8f acc[NB / 16] = {};

#pragma unroll
    for (int kb = 0; kb < GK; kb += 32) {
        if (kb + 32 < GK) __builtin_prefetch(arow + kb + 32, 0, 1);

        const float4* ap0 = reinterpret_cast<const float4*>(arow + kb + koff);
        const float4* ap1 = reinterpret_cast<const float4*>(arow + kb + 8 + koff);
        float4 a0 = ap0[0], a1 = ap0[1];
        float4 a2 = ap1[0], a3 = ap1[1];

        v16bf af;
        af[0]  = f32_to_bf16(a0.x); af[1]  = f32_to_bf16(a0.y);
        af[2]  = f32_to_bf16(a0.z); af[3]  = f32_to_bf16(a0.w);
        af[4]  = f32_to_bf16(a1.x); af[5]  = f32_to_bf16(a1.y);
        af[6]  = f32_to_bf16(a1.z); af[7]  = f32_to_bf16(a1.w);
        af[8]  = f32_to_bf16(a2.x); af[9]  = f32_to_bf16(a2.y);
        af[10] = f32_to_bf16(a2.z); af[11] = f32_to_bf16(a2.w);
        af[12] = f32_to_bf16(a3.x); af[13] = f32_to_bf16(a3.y);
        af[14] = f32_to_bf16(a3.z); af[15] = f32_to_bf16(a3.w);

        const int cs = ((((kb + kofB) >> 4) ^ ln16) << 4);  // swizzled chunk
#pragma unroll
        for (int t = 0; t < NB / 16; ++t) {
            const int n = t * 16 + ln16;
            v16bf bfv = *reinterpret_cast<const v16bf*>(&Blds[n * GK + cs]);
            acc[t] = __builtin_amdgcn_wmma_f32_16x16x32_bf16(
                /*neg_a=*/false, af, /*neg_b=*/false, bfv,
                /*c_mod=*/(short)0, acc[t], /*reuse_a=*/false, /*reuse_b=*/false);
        }
    }

    // ---- epilogue: C layout VGPR r -> M = r (lanes 0-15) / 8+r (lanes 16-31)
    const int rb = (lane >> 4) << 3;
#pragma unroll
    for (int t = 0; t < NB / 16; ++t) {
        const int col = col0 + t * 16 + ln16;
        const float bv = bias ? bias[col] : 0.0f;
#pragma unroll
        for (int r = 0; r < 8; ++r) {
            size_t idx = (size_t)(row0 + rb + r) * N + col;
            float val = acc[t][r] + bv;
            if (resid) val += resid[idx];
            C[idx] = val;
        }
    }
}

// ---------------------------------------------------------------------------
// Kernel 3: deformable sampling.  One wave32 per (query, head); lane = channel.
// value layout (b, y, x, level, head, 32ch) -> every corner gather is one
// coalesced 128-byte wave load.
// ---------------------------------------------------------------------------
__global__ void msda_sample_kernel(const float* __restrict__ off,
                                   const float* __restrict__ attn,
                                   const float* __restrict__ value,
                                   const float* __restrict__ ref_xy,
                                   const int*   __restrict__ batch_offsets,
                                   const int*   __restrict__ spatial_shapes,
                                   float* __restrict__ out) {
    int wid  = (blockIdx.x * blockDim.x + threadIdx.x) >> 5;
    int lane = threadIdx.x & 31;
    if (wid >= NQ * HEADS) return;
    int n = wid >> 3;
    int h = wid & 7;

    int b = 0;
#pragma unroll
    for (int i = 1; i <= BB; ++i)
        if (n >= batch_offsets[i]) b = i;

    float refx = ref_xy[n * 2 + 0];
    float refy = ref_xy[n * 2 + 1];

    // softmax over the 16 (level,point) logits for this head
    const float* ap = attn + (size_t)n * (HEADS * 16) + h * 16;
    float w[16], mx = -3.4e38f;
#pragma unroll
    for (int j = 0; j < 16; ++j) { w[j] = ap[j]; mx = fmaxf(mx, w[j]); }
    float s = 0.f;
#pragma unroll
    for (int j = 0; j < 16; ++j) { w[j] = __expf(w[j] - mx); s += w[j]; }
    float inv = 1.0f / s;

    const float* op = off + (size_t)n * (HEADS * 32) + h * 32;
    float acc = 0.f;

#pragma unroll
    for (int l = 0; l < LEVELS; ++l) {
        int Hl = spatial_shapes[l * 2 + 0];
        int Wl = spatial_shapes[l * 2 + 1];
        float fH = (float)Hl, fW = (float)Wl;
#pragma unroll
        for (int p = 0; p < POINTS; ++p) {
            float ox = op[(l * POINTS + p) * 2 + 0];
            float oy = op[(l * POINTS + p) * 2 + 1];
            float x = refx * fW + ox - 0.5f;
            float y = refy * fH + oy - 0.5f;
            float fx0 = floorf(x), fy0 = floorf(y);
            int x0 = (int)fx0, y0 = (int)fy0;
            float fx = x - fx0, fy = y - fy0;
            float aw = w[l * POINTS + p] * inv;

            float csum = 0.f;
#pragma unroll
            for (int c = 0; c < 4; ++c) {
                int yi = y0 + (c >> 1);
                int xi = x0 + (c & 1);
                float wy = (c >> 1) ? fy : (1.f - fy);
                float wx = (c & 1)  ? fx : (1.f - fx);
                bool valid = (yi >= 0) && (yi < Hl) && (xi >= 0) && (xi < Wl);
                int yc = min(max(yi, 0), HMAX - 1);
                int xc = min(max(xi, 0), WMAX - 1);
                size_t idx = ((((size_t)((b * HMAX + yc) * WMAX + xc)) * LEVELS + l)
                              * HEADS + h) * HDIM + lane;
                float v = value[idx];
                csum += valid ? wy * wx * v : 0.f;
            }
            acc += aw * csum;
        }
    }
    out[(size_t)n * DD + h * HDIM + lane] = acc;
}

// ---------------------------------------------------------------------------
extern "C" void kernel_launch(void* const* d_in, const int* in_sizes, int n_in,
                              void* d_out, int out_size, void* d_ws, size_t ws_size,
                              hipStream_t stream) {
    const float* queries   = (const float*)d_in[0];
    const float* query_pos = (const float*)d_in[1];
    const float* ref_xy    = (const float*)d_in[2];
    const int*   batch_off = (const int*)  d_in[3];
    const float* fmaps     = (const float*)d_in[4];
    const int*   shapes    = (const int*)  d_in[5];
    const float* ln_gamma  = (const float*)d_in[6];
    const float* ln_beta   = (const float*)d_in[7];
    const float* W_value   = (const float*)d_in[8];
    const float* b_value   = (const float*)d_in[9];
    const float* W_off     = (const float*)d_in[10];
    const float* b_off     = (const float*)d_in[11];
    const float* W_attn    = (const float*)d_in[12];
    const float* b_attn    = (const float*)d_in[13];
    const float* W_out     = (const float*)d_in[14];
    const float* b_out     = (const float*)d_in[15];
    float* out = (float*)d_out;

    // workspace carve-up (floats)
    float* q_buf     = (float*)d_ws;                       // NQ*256
    float* off_buf   = q_buf    + (size_t)NQ * DD;         // NQ*256
    float* attn_buf  = off_buf  + (size_t)NQ * DD;         // NQ*128
    float* msda_buf  = attn_buf + (size_t)NQ * (HEADS*16); // NQ*256
    float* value_buf = msda_buf + (size_t)NQ * DD;         // 131072*256

    const int MV = BB * HMAX * WMAX * LEVELS;  // 131072 value rows

    // 1) layernorm
    ln_kernel<<<NQ / 8, 256, 0, stream>>>(queries, query_pos, ln_gamma, ln_beta,
                                          q_buf, NQ);

    // 2) value projection: fmaps(MVx256) @ W_value + b_value
    wmma_gemm_bf16<<<(MV / 128) * (DD / NB), 256, 0, stream>>>(
        fmaps, W_value, b_value, nullptr, value_buf, MV, DD);

    // 3) sampling offsets: q @ W_off + b_off   (N=256)
    wmma_gemm_bf16<<<(NQ / 128) * (DD / NB), 256, 0, stream>>>(
        q_buf, W_off, b_off, nullptr, off_buf, NQ, DD);

    // 4) attention logits: q @ W_attn + b_attn  (N=128)
    wmma_gemm_bf16<<<(NQ / 128) * (128 / NB), 256, 0, stream>>>(
        q_buf, W_attn, b_attn, nullptr, attn_buf, NQ, 128);

    // 5) softmax + bilinear deformable sampling
    msda_sample_kernel<<<(NQ * HEADS) / 8, 256, 0, stream>>>(
        off_buf, attn_buf, value_buf, ref_xy, batch_off, shapes, msda_buf);

    // 6) output projection + residual: msda @ W_out + b_out + queries
    wmma_gemm_bf16<<<(NQ / 128) * (DD / NB), 256, 0, stream>>>(
        msda_buf, W_out, b_out, queries, out, NQ, DD);
}
